// NormalAttention_26482768347189
// MI455X (gfx1250) — matrix operand order repaired and checked by reading, not compiled, not run on hardware
//
#include <hip/hip_runtime.h>
#include <hip/hip_bf16.h>
#include <math.h>

// ---------------------------------------------------------------------------
// Problem constants (from reference): B=32, S=2048, D_IN=D_TAR=H=1024
// ---------------------------------------------------------------------------
#define BB   32
#define SS   2048
#define HH   1024
#define MT   64          // rows of flattened [B*S] per workgroup in main GEMM

#define LDS_A_BYTES (MT * HH * 2)       // 131072: A panel, bf16
#define LDS_B_BYTES (128 * 32 * 2)      // 8192:   one B tile [128 n][32 k] bf16
#define LDS_B0      (LDS_A_BYTES)
#define LDS_B1      (LDS_A_BYTES + LDS_B_BYTES)
#define LDS_LG      (LDS_A_BYTES + 2 * LDS_B_BYTES)
#define LDS_TOTAL   (LDS_LG + 8 * MT * 4)

typedef __attribute__((ext_vector_type(16))) __bf16    v16bf;
typedef __attribute__((ext_vector_type(8)))  float     v8f;
typedef __attribute__((ext_vector_type(4)))  unsigned  v4u;
typedef __attribute__((ext_vector_type(8)))  int       v8i_;
typedef __attribute__((ext_vector_type(4)))  int       v4i_;

union FragAB { v16bf v; unsigned u[8]; };
union FragC  { v8f   v; float    f[8]; };

__device__ __forceinline__ unsigned short f2bf(float f) {
  union { float f; unsigned u; } x; x.f = f;
  unsigned u = x.u;
  unsigned r = u + 0x7FFFu + ((u >> 16) & 1u);   // round-to-nearest-even
  return (unsigned short)(r >> 16);
}

__device__ __forceinline__ float fast_tanh(float z) {
  // tanh(z) = 1 - 2/(exp(2z)+1); robust at +/-inf via __expf saturation
  return 1.0f - 2.0f / (__expf(2.0f * z) + 1.0f);
}

// ---------------------------------------------------------------------------
// TDM: async-load one B tile [tile_dim1=128 rows][tile_dim0=32 elem of 2B]
// from WeffT (row stride 1024 elem) into LDS at lds_off. D# per ISA 8.3/8.4.
// ---------------------------------------------------------------------------
__device__ __forceinline__ void tdm_load_tile(const unsigned short* gsrc,
                                              unsigned lds_off) {
  unsigned long long ga = (unsigned long long)(uintptr_t)gsrc;
  v4u g0;
  g0[0] = 1u;                                           // count=1, user mode
  g0[1] = lds_off;                                      // lds_addr (bytes)
  g0[2] = (unsigned)(ga & 0xFFFFFFFFu);                 // global_addr[31:0]
  g0[3] = ((unsigned)(ga >> 32) & 0x01FFFFFFu)          // global_addr[56:32]
          | 0x80000000u;                                // type=2 ("image")
  v8i_ g1;
  g1[0] = 0x00010000;          // workgroup_mask=0 | data_size=1 (2 bytes)
  g1[1] = (int)(1024u << 16);  // atomic_barrier_addr=0 | tensor_dim0.lo=1024
  g1[2] = (int)(1024u << 16);  // tensor_dim0.hi=0      | tensor_dim1.lo=1024
  g1[3] = (int)(32u << 16);    // tensor_dim1.hi=0      | tile_dim0=32
  g1[4] = 128;                 // tile_dim1=128         | tile_dim2=0
  g1[5] = 1024;                // tensor_dim0_stride[31:0]=1024
  g1[6] = 0;                   // stride.hi=0 | tensor_dim1_stride.lo=0 (unused, 2D)
  g1[7] = 0;
  v4i_ z4 = {0, 0, 0, 0};
#if __clang_major__ >= 23
  v8i_ z8 = {0, 0, 0, 0, 0, 0, 0, 0};
  __builtin_amdgcn_tensor_load_to_lds(g0, g1, z4, z4, z8, 0);
#else
  __builtin_amdgcn_tensor_load_to_lds(g0, g1, z4, z4, 0);
#endif
}

// ---------------------------------------------------------------------------
// Precompute 1: Weff[k][n] = sum_j W_in[k][j] * W1[j][n]  (W1_top), bf16,
// k-major (coalesced over n). grid (1024, 4) x 256
// ---------------------------------------------------------------------------
__global__ void __launch_bounds__(256)
k_weff(const float* __restrict__ W_in, const float* __restrict__ W1,
       unsigned short* __restrict__ Weff) {
  const int k = blockIdx.x;
  const int n = blockIdx.y * 256 + threadIdx.x;
  const float* __restrict__ wr = W_in + k * HH;   // uniform (scalar) stream
  float acc = 0.0f;
#pragma unroll 4
  for (int j = 0; j < HH; ++j)
    acc = fmaf(wr[j], W1[j * HH + n], acc);
  Weff[k * HH + n] = f2bf(acc);
}

// ---------------------------------------------------------------------------
// Precompute 1b: WeffT[n][k] = Weff[k][n]  (LDS-tiled 64x64, both sides
// coalesced). grid (16,16) x 256. Makes the GEMM's B tiles plain 2D blocks
// so the TDM can stream them with no in-loop transpose.
// ---------------------------------------------------------------------------
__global__ void __launch_bounds__(256)
k_transpose(const unsigned short* __restrict__ src,
            unsigned short* __restrict__ dst) {
  __shared__ unsigned short t[64][65];              // +1 pad: no bank conflicts
  const int kt = blockIdx.x * 64, nt = blockIdx.y * 64;
  for (int i = threadIdx.x; i < 64 * 64; i += 256) {
    const int r = i >> 6, c = i & 63;
    t[r][c] = src[(size_t)(kt + r) * HH + nt + c];
  }
  __syncthreads();
  for (int i = threadIdx.x; i < 64 * 64; i += 256) {
    const int r = i >> 6, c = i & 63;
    dst[(size_t)(nt + r) * HH + kt + c] = t[c][r];
  }
}

// ---------------------------------------------------------------------------
// Precompute 2: ct[b][k] = target[b]@W_tar[:,k] + b_tar[k]   grid (32,4) x 256
// ---------------------------------------------------------------------------
__global__ void __launch_bounds__(256)
k_ct(const float* __restrict__ target, const float* __restrict__ W_tar,
     const float* __restrict__ b_tar, float* __restrict__ ct) {
  const int b = blockIdx.x;
  const int k = blockIdx.y * 256 + threadIdx.x;
  const float* __restrict__ t = target + b * HH;
  float acc = b_tar[k];
#pragma unroll 4
  for (int j = 0; j < HH; ++j)
    acc = fmaf(t[j], W_tar[j * HH + k], acc);
  ct[b * HH + k] = acc;
}

// ---------------------------------------------------------------------------
// Precompute 3: tbias[b][h] = b1[h] + b_in@W1_top[:,h] + ct[b]@W1_bot[:,h]
// grid (32,4) x 256
// ---------------------------------------------------------------------------
__global__ void __launch_bounds__(256)
k_tbias(const float* __restrict__ ct, const float* __restrict__ W1,
        const float* __restrict__ b_in, const float* __restrict__ b1,
        float* __restrict__ tbias) {
  const int b = blockIdx.x;
  const int h = blockIdx.y * 256 + threadIdx.x;
  float acc = b1[h];
#pragma unroll 4
  for (int j = 0; j < HH; ++j)
    acc = fmaf(b_in[j], W1[j * HH + h], acc);
  const float* __restrict__ c = ct + b * HH;
#pragma unroll 4
  for (int k = 0; k < HH; ++k)
    acc = fmaf(c[k], W1[(HH + k) * HH + h], acc);
  tbias[b * HH + h] = acc;
}

// ---------------------------------------------------------------------------
// Main fused kernel: logits[row] = sum_n tanh(X[row]@Weff[:,n] + tbias[b,n])*Wc[n]
// 256 threads = 8 waves, MT=64 rows/WG. A panel (bf16) lives in LDS for the
// whole block. B tiles ([128 n][32 k] of WeffT) are TDM double-buffered:
// wave 0 issues tensor_load_to_lds for tile t+1, s_wait_tensorcnt(1) ensures
// tile t complete (TDM is in-order per wave), barrier, all waves WMMA.
// NOTE: descriptor LDS addresses assume dynamic LDS starts at offset 0
// (this kernel has no static __shared__).
// ---------------------------------------------------------------------------
__global__ void __launch_bounds__(256)
k_gemm_logits(const float* __restrict__ X, const unsigned short* __restrict__ WeffT,
              const float* __restrict__ tbias, const float* __restrict__ Wc,
              float* __restrict__ logits) {
  extern __shared__ char smem[];
  unsigned short* Ald = (unsigned short*)smem;            // [MT][1024] bf16
  float* Lg = (float*)(smem + LDS_LG);                    // [8 waves][MT]

  const int tid  = threadIdx.x;
  const int wave = tid >> 5;
  const int lane = tid & 31;
  const int lh   = lane & 15;        // C: N = lh ; A/B: row/col = lh
  const int hi   = lane >> 4;        // half selector (K-group / M+8)
  const long r0  = (long)blockIdx.x * MT;
  const int b    = (int)(r0 >> 11);  // 2048 rows per batch, MT | 2048

  // Kick off tile 0 immediately (overlaps with A staging).
  if (wave == 0) tdm_load_tile(WeffT, LDS_B0);

  // zero per-wave logit partials
  for (int i = tid; i < 8 * MT; i += 256) Lg[i] = 0.0f;

  // Stage A panel once: f32 -> bf16 into LDS (coalesced float4 loads)
  const float4* __restrict__ Xv = (const float4*)(X + r0 * HH);
  for (int i = tid; i < MT * HH / 4; i += 256) {
    float4 f = Xv[i];
    unsigned lo = (unsigned)f2bf(f.x) | ((unsigned)f2bf(f.y) << 16);
    unsigned hw = (unsigned)f2bf(f.z) | ((unsigned)f2bf(f.w) << 16);
    ((uint2*)Ald)[i] = make_uint2(lo, hw);
  }
  __syncthreads();

  for (int nc = 0; nc < 8; ++nc) {           // 8 N-chunks of 128 cols
    const int nbase = nc * 128;
    const int ncol  = nbase + wave * 16 + lh;

    FragC acc[4];
#pragma unroll
    for (int m = 0; m < 4; ++m) acc[m].v = (v8f){0, 0, 0, 0, 0, 0, 0, 0};

    for (int kb = 0; kb < HH; kb += 32) {    // K loop, steps of 32
      const int t = nc * 32 + (kb >> 5);     // flat tile index (0..255)
      if (wave == 0) {
        const int t1 = t + 1;
        if (t1 < 256) {                      // prefetch next tile (ping-pong)
          const int nc2 = t1 >> 5, kb2 = (t1 & 31) << 5;
          tdm_load_tile(WeffT + (size_t)(nc2 * 128) * HH + kb2,
                        (t1 & 1) ? LDS_B1 : LDS_B0);
          __builtin_amdgcn_s_wait_tensorcnt(1);  // in-order => tile t done
        } else {
          __builtin_amdgcn_s_wait_tensorcnt(0);  // last tile: drain
        }
      }
      __syncthreads();                       // tile t visible to all waves
      // ping-pong buffer select via byte offset (no LDS pointer arrays:
      // addrspacecast in a static initializer breaks the device linker)
      const unsigned short* Bcur =
          (const unsigned short*)(smem + ((t & 1) ? LDS_B1 : LDS_B0));

      // B fragment: lane -> n = lh, half -> K-group(8), VGPR v -> k=hi*8+(v<4?2v:16+2(v-4)),+1
      FragAB bf;
      const int nloc = wave * 16 + lh;
#pragma unroll
      for (int v = 0; v < 8; ++v)
        bf.u[v] = *(const unsigned*)(Bcur + nloc * 32 + hi * 8 + (v < 4 ? 2 * v : 16 + 2 * (v - 4)));

      // Hoist all 4 A fragments so their ds_loads batch ahead of the WMMAs
      FragAB af[4];
#pragma unroll
      for (int m = 0; m < 4; ++m) {
        const int row = m * 16 + lh;
#pragma unroll
        for (int v = 0; v < 8; ++v) {
          const int k = kb + hi * 8 + (v < 4 ? 2 * v : 16 + 2 * (v - 4));
          af[m].u[v] = *(const unsigned*)(Ald + row * HH + k);
        }
      }
#pragma unroll
      for (int m = 0; m < 4; ++m)
        acc[m].v = __builtin_amdgcn_wmma_f32_16x16x32_bf16(
            false, af[m].v, false, bf.v, (short)0, acc[m].v, false, false);

      __syncthreads();   // all waves done with buffer (t&1) before t+2 lands
    }

    // Epilogue: z += tbias, tanh, *Wc, reduce the wave's 16 columns
    const float tb  = tbias[b * HH + ncol];
    const float wsc = Wc[ncol];
#pragma unroll
    for (int m = 0; m < 4; ++m) {
#pragma unroll
      for (int v = 0; v < 8; ++v) {
        float c = fast_tanh(acc[m].f[v] + tb) * wsc;
        c += __shfl_xor(c, 1, 32);
        c += __shfl_xor(c, 2, 32);
        c += __shfl_xor(c, 4, 32);
        c += __shfl_xor(c, 8, 32);           // lanes 0 / 16 hold 16-col sums
        if (lh == 0)                          // per-wave slice: deterministic
          Lg[wave * MT + m * 16 + v + 8 * hi] += c;
      }
    }
  }
  __syncthreads();
  if (tid < MT) {
    float s = 0.0f;
#pragma unroll
    for (int w = 0; w < 8; ++w) s += Lg[w * MT + tid];
    logits[r0 + tid] = s;                    // b_c omitted: softmax-invariant
  }
}

// ---------------------------------------------------------------------------
// Softmax over S + out[b,d] = sum_s attn[b,s] * X[b,s,d]
// grid (32, 4) x 256 ; each block recomputes softmax (logits are tiny) and
// accumulates 256 columns of the 256 MB X stream (the HBM-bound part).
// ---------------------------------------------------------------------------
__global__ void __launch_bounds__(256)
k_softmax_out(const float* __restrict__ logits, const float* __restrict__ X,
              float* __restrict__ out) {
  __shared__ float attn[SS];
  __shared__ float red[256];
  const int b = blockIdx.x, dc = blockIdx.y, tid = threadIdx.x;
  const float* __restrict__ lg = logits + b * SS;

  float mx = -3.4e38f;
  for (int s = tid; s < SS; s += 256) { float v = lg[s]; attn[s] = v; mx = fmaxf(mx, v); }
  red[tid] = mx; __syncthreads();
  for (int off = 128; off > 0; off >>= 1) {
    if (tid < off) red[tid] = fmaxf(red[tid], red[tid + off]);
    __syncthreads();
  }
  mx = red[0]; __syncthreads();

  float sm = 0.0f;
  for (int s = tid; s < SS; s += 256) { float e = __expf(attn[s] - mx); attn[s] = e; sm += e; }
  red[tid] = sm; __syncthreads();
  for (int off = 128; off > 0; off >>= 1) {
    if (tid < off) red[tid] += red[tid + off];
    __syncthreads();
  }
  const float inv = 1.0f / red[0];

  const int d = dc * 256 + tid;
  const float* __restrict__ Xb = X + (long)b * SS * HH + d;
  float acc = 0.0f;
#pragma unroll 8
  for (int s = 0; s < SS; ++s)
    acc = fmaf(attn[s], Xb[(long)s * HH], acc);    // coalesced 1KB segments
  out[b * HH + d] = acc * inv;
}

// ---------------------------------------------------------------------------
// Launch: inputs in setup_inputs() order, all float32.
// ws: Weff(2MB) | WeffT(2MB) | ct(128KB) | tbias(128KB) | logits(256KB)
// ---------------------------------------------------------------------------
extern "C" void kernel_launch(void* const* d_in, const int* in_sizes, int n_in,
                              void* d_out, int out_size, void* d_ws, size_t ws_size,
                              hipStream_t stream) {
  (void)in_sizes; (void)n_in; (void)out_size; (void)ws_size;
  const float* X      = (const float*)d_in[0];   // [32,2048,1024]
  const float* target = (const float*)d_in[1];   // [32,1024]
  const float* W_in   = (const float*)d_in[2];   // [1024,1024]
  const float* b_in   = (const float*)d_in[3];   // [1024]
  const float* W_tar  = (const float*)d_in[4];   // [1024,1024]
  const float* b_tar  = (const float*)d_in[5];   // [1024]
  const float* W1     = (const float*)d_in[6];   // [2048,1024]
  const float* b1     = (const float*)d_in[7];   // [1024]
  const float* W_c    = (const float*)d_in[8];   // [1024,1]
  // d_in[9] = b_c : softmax-invariant, unused.

  char* ws = (char*)d_ws;
  unsigned short* Weff  = (unsigned short*)(ws);
  unsigned short* WeffT = (unsigned short*)(ws + 2u * 1024u * 1024u);
  float*          ct    = (float*)(ws + 4u * 1024u * 1024u);
  float*          tbias = (float*)(ws + 4u * 1024u * 1024u + 128u * 1024u);
  float*          lgts  = (float*)(ws + 4u * 1024u * 1024u + 256u * 1024u);
  float*          out   = (float*)d_out;         // [32,1,1024]

  k_weff     <<<dim3(1024, 4), 256, 0, stream>>>(W_in, W1, Weff);
  k_transpose<<<dim3(16, 16),  256, 0, stream>>>(Weff, WeffT);
  k_ct       <<<dim3(BB, 4),   256, 0, stream>>>(target, W_tar, b_tar, ct);
  k_tbias    <<<dim3(BB, 4),   256, 0, stream>>>(ct, W1, b_in, b1, tbias);

  k_gemm_logits<<<dim3((BB * SS) / MT), 256, LDS_TOTAL, stream>>>(X, WeffT, tbias, W_c, lgts);

  k_softmax_out<<<dim3(BB, 4), 256, 0, stream>>>(lgts, X, out);
}
